// TransformerModel_77034533421082
// MI455X (gfx1250) — matrix operand order repaired and checked by reading, not compiled, not run on hardware
//
#include <hip/hip_runtime.h>
#include <hip/hip_bf16.h>
#include <math.h>

// ---------------------------------------------------------------------------
// Model dims (fixed by the reference)
// ---------------------------------------------------------------------------
#define LAYERS 6
#define DMODEL 1024
#define NHEAD  16
#define DFFN   4096
#define VOCAB  32000
#define BATCH  2
#define SEQ    1024
#define DEPTH  64          // DMODEL / NHEAD
#define NTOK   (BATCH*SEQ) // 2048
#define EPS    1e-6f

typedef __attribute__((ext_vector_type(16))) __bf16        v16bf;
typedef __attribute__((ext_vector_type(8)))  float         v8f;
typedef __attribute__((ext_vector_type(4)))  unsigned int  v4u;
typedef __attribute__((ext_vector_type(8)))  unsigned int  v8u;

// ---------------------------------------------------------------------------
// WMMA helper: D = A(16x32 bf16) * B(32x16 bf16) + C(16x16 f32)
// ---------------------------------------------------------------------------
__device__ __forceinline__ v8f wmma_bf16(v16bf a, v16bf b, v8f c) {
    return __builtin_amdgcn_wmma_f32_16x16x32_bf16(
        /*neg_a=*/false, a, /*neg_b=*/false, b,
        /*c_mod=*/(short)0, c, /*reuse_a=*/false, /*reuse_b=*/false);
}

// A-matrix fragment (16x32, MxK) per ISA layout (§7.12.2):
//   lanes 0-15: M=lane,  K chunks [k0, k0+8) and [k0+16, k0+24)
//   lanes16-31: M=l-16,  K chunks [k0+8,k0+16) and [k0+24, k0+32)
__device__ __forceinline__ v16bf load_frag_a(const __bf16* base, int ld,
                                             int mBase, int k0) {
    int lane = threadIdx.x & 31;
    int m  = mBase + (lane & 15);
    int kb = k0 + ((lane >> 4) << 3);
    union { uint4 u[2]; v16bf v; } r;
    r.u[0] = *(const uint4*)(base + (size_t)m * ld + kb);
    r.u[1] = *(const uint4*)(base + (size_t)m * ld + kb + 16);
    return r.v;
}

// B-matrix fragment (32x16, KxN); LDS holds B transposed: bt[n][k] = B[k][n]
//   lanes 0-15: N=lane,  K chunks [k0, k0+8) and [k0+16, k0+24);  lanes16-31: N=l-16, K chunks [k0+8,k0+16) and [k0+24,k0+32)
__device__ __forceinline__ v16bf load_frag_b(const __bf16* bt, int ld,
                                             int nBase, int k0) {
    int lane = threadIdx.x & 31;
    int n  = nBase + (lane & 15);
    int kb = k0 + ((lane >> 4) << 3);
    union { uint4 u[2]; v16bf v; } r;
    r.u[0] = *(const uint4*)(bt + (size_t)n * ld + kb);
    r.u[1] = *(const uint4*)(bt + (size_t)n * ld + kb + 16);
    return r.v;
}

// ---------------------------------------------------------------------------
// Tensor Data Mover: one 2D bf16 tile load (global -> LDS) with LDS row
// padding done by the TDM (D# pad_interval/pad_amount), per ISA §8.
//   tile = tile1 rows x tile0 elements, source row stride = stride0 elements.
// ---------------------------------------------------------------------------
__device__ __forceinline__ void tdm_load_tile(unsigned lds_off, const __bf16* g,
                                              unsigned tdim0, unsigned tdim1,
                                              unsigned long long stride0,
                                              unsigned tile0, unsigned tile1,
                                              unsigned padI, unsigned padA) {
    unsigned long long ga = (unsigned long long)(size_t)g;
    v4u g0;
    g0[0] = 1u;                                     // count=1, user mode
    g0[1] = lds_off;                                // lds_addr (bytes)
    g0[2] = (unsigned)ga;                           // global_addr[31:0]
    g0[3] = (unsigned)((ga >> 32) & 0x01ffffffu)    // global_addr[56:32]
            | (2u << 30);                           // type = 2 ("image")
    v8u g1;
    g1[0] = (1u << 16)                              // data_size = 2 bytes
          | (1u << 20)                              // pad_enable
          | (padI << 22) | (padA << 25);            // pad interval/amount codes
    g1[1] = (tdim0 & 0xffffu) << 16;                // tensor_dim0[15:0]
    g1[2] = (tdim0 >> 16) | ((tdim1 & 0xffffu) << 16);
    g1[3] = (tdim1 >> 16) | (tile0 << 16);          // tile_dim0
    g1[4] = tile1;                                  // tile_dim1 (tile_dim2 = 0)
    g1[5] = (unsigned)stride0;                      // tensor_dim0_stride[31:0]
    g1[6] = (unsigned)((stride0 >> 32) & 0xffffu);  // stride[47:32]
    g1[7] = 0u;
    asm volatile("tensor_load_to_lds %0, %1" :: "s"(g0), "s"(g1) : "memory");
}

// ---------------------------------------------------------------------------
// Generic GEMM: C[M,N] = act(A[M,K] * B[K,N] + bias[N])
//   A: bf16 [M,K] row-major.  Wt: bf16 [N,K] (B pre-transposed).
//   Outputs: optional fp32 Cf and/or bf16 Cb.
// Double-buffered TDM staging: tile i+1 streams into the alternate LDS buffer
// while the waves run WMMA on tile i (TENSORcnt completes in order, so
// s_wait_tensorcnt <= 2 after issuing 2 more ops proves tile i has landed).
// Block 256 thr (8 waves), tile 128x128, BK=32, wave -> 32x64 (2x4 frags).
// ---------------------------------------------------------------------------
#define GBM 128
#define GBN 128
#define GBK 32
#define GLD 40   // padded LDS row stride = 16 DW data + 4 DW pad (codes 3/3)

__global__ __launch_bounds__(256)
void gemm_tdm_kernel(const __bf16* __restrict__ A, const __bf16* __restrict__ Wt,
                     const float* __restrict__ bias,
                     float* __restrict__ Cf, __bf16* __restrict__ Cb,
                     int M, int N, int K, int relu) {
    __shared__ alignas(16) __bf16 As[2][GBM * GLD];
    __shared__ alignas(16) __bf16 Bs[2][GBN * GLD];

    const int t    = threadIdx.x;
    const int wid  = t >> 5;
    const int lane = t & 31;
    const int m0   = blockIdx.y * GBM;
    const int n0   = blockIdx.x * GBN;
    const int wm   = (wid >> 1) * 32;
    const int wn   = (wid & 1) * 64;

    v8f acc[2][4];
    const v8f vzero = {0.f,0.f,0.f,0.f,0.f,0.f,0.f,0.f};
    for (int i = 0; i < 2; ++i)
        for (int j = 0; j < 4; ++j) acc[i][j] = vzero;

    const int KT = K / GBK;

    // pre-issue tile 0 into buffer 0
    if (wid == 0) {
        tdm_load_tile((unsigned)(size_t)(void*)&As[0][0],
                      A + (size_t)m0 * K,
                      (unsigned)K, (unsigned)M, (unsigned long long)K,
                      GBK, GBM, /*padI=*/3, /*padA=*/3);
        tdm_load_tile((unsigned)(size_t)(void*)&Bs[0][0],
                      Wt + (size_t)n0 * K,
                      (unsigned)K, (unsigned)N, (unsigned long long)K,
                      GBK, GBN, /*padI=*/3, /*padA=*/3);
    }

    for (int kt = 0; kt < KT; ++kt) {
        const int cur = kt & 1;
        __syncthreads();          // alternate buffer's readers are done
        if (wid == 0) {
            if (kt + 1 < KT) {    // stream next tile while computing this one
                const int k0n = (kt + 1) * GBK;
                tdm_load_tile((unsigned)(size_t)(void*)&As[cur ^ 1][0],
                              A + (size_t)m0 * K + k0n,
                              (unsigned)K, (unsigned)M, (unsigned long long)K,
                              GBK, GBM, 3, 3);
                tdm_load_tile((unsigned)(size_t)(void*)&Bs[cur ^ 1][0],
                              Wt + (size_t)n0 * K + k0n,
                              (unsigned)K, (unsigned)N, (unsigned long long)K,
                              GBK, GBN, 3, 3);
                __builtin_amdgcn_s_wait_tensorcnt(2);  // tile kt complete
            } else {
                __builtin_amdgcn_s_wait_tensorcnt(0);  // last tile complete
            }
        }
        __syncthreads();          // tile kt visible to all waves

        v16bf a[2], b[4];
        for (int i = 0; i < 2; ++i) a[i] = load_frag_a(&As[cur][0], GLD, wm + i * 16, 0);
        for (int j = 0; j < 4; ++j) b[j] = load_frag_b(&Bs[cur][0], GLD, wn + j * 16, 0);
        for (int i = 0; i < 2; ++i)
            for (int j = 0; j < 4; ++j)
                acc[i][j] = wmma_bf16(a[i], b[j], acc[i][j]);
    }

    // epilogue (C layout: lanes 0-15 N=lane / M=r ; lanes 16-31 N=l-16 / M=r+8)
    const int cn = lane & 15;
    const int rm = (lane >> 4) << 3;
    for (int i = 0; i < 2; ++i)
        for (int j = 0; j < 4; ++j) {
            int gn = n0 + wn + j * 16 + cn;
            float bsv = bias[gn];
            for (int r = 0; r < 8; ++r) {
                int gm = m0 + wm + i * 16 + rm + r;
                float vv = acc[i][j][r] + bsv;
                if (relu) vv = fmaxf(vv, 0.0f);
                size_t idx = (size_t)gm * N + gn;
                if (Cf) Cf[idx] = vv;
                if (Cb) Cb[idx] = (__bf16)vv;
            }
        }
}

// ---------------------------------------------------------------------------
// Weight prep: Wt[n][k] = (bf16)W[k][n]   (32x32 LDS-tiled transpose+convert)
// ---------------------------------------------------------------------------
__global__ __launch_bounds__(256)
void transpose_bf16_kernel(const float* __restrict__ W, __bf16* __restrict__ Wt,
                           int K, int N) {
    __shared__ float tile[32][33];
    const int n0 = blockIdx.x * 32, k0 = blockIdx.y * 32;
    const int tx = threadIdx.x & 31, ty = threadIdx.x >> 5;
    for (int i = 0; i < 4; ++i) {
        int k = ty + i * 8;
        tile[k][tx] = W[(size_t)(k0 + k) * N + n0 + tx];
    }
    __syncthreads();
    for (int i = 0; i < 4; ++i) {
        int n = ty + i * 8;
        Wt[(size_t)(n0 + n) * K + k0 + tx] = (__bf16)tile[tx][n];
    }
}

// ---------------------------------------------------------------------------
// Flash-style attention, bf16 Q/K/V [NTOK, DMODEL] (heads in columns).
// Block 128 thr (4 waves) = 64 q-rows of one (b,h); Q/K tiles TDM-staged.
// ---------------------------------------------------------------------------
#define ALD 72   // 32 DW data + 4 DW pad per row (codes 4/3), 144B aligned

__global__ __launch_bounds__(128)
void attention_kernel(const __bf16* __restrict__ Q, const __bf16* __restrict__ K,
                      const __bf16* __restrict__ V, __bf16* __restrict__ O) {
    __shared__ alignas(16) __bf16 Qs[64 * ALD];
    __shared__ alignas(16) __bf16 Ks[64 * ALD];
    __shared__ alignas(16) __bf16 Vt[64 * ALD];        // Vt[d][key]
    __shared__ alignas(16) __bf16 Ps[4 * 16 * ALD];    // per-wave P patch

    const int t    = threadIdx.x;
    const int wid  = t >> 5;
    const int lane = t & 31;
    const int qblk = blockIdx.x;            // 0..15
    const int bh   = blockIdx.y;            // 0..31
    const int b    = bh >> 4;
    const int h    = bh & 15;
    const size_t rowbase = (size_t)b * SEQ;
    const int    cbase   = h * DEPTH;

    // ---- stage Q tile (64x64) via TDM
    if (wid == 0) {
        tdm_load_tile((unsigned)(size_t)(void*)&Qs[0],
                      Q + (rowbase + qblk * 64) * DMODEL + cbase,
                      DMODEL, NTOK, DMODEL, 64, 64, /*padI=*/4, /*padA=*/3);
        __builtin_amdgcn_s_wait_tensorcnt(0);
    }
    __syncthreads();

    v16bf qa0 = load_frag_a(Qs, ALD, wid * 16, 0);
    v16bf qa1 = load_frag_a(Qs, ALD, wid * 16, 32);

    const v8f vzero = {0.f,0.f,0.f,0.f,0.f,0.f,0.f,0.f};
    v8f oacc[4];
    for (int n = 0; n < 4; ++n) oacc[n] = vzero;
    float mrow[8], lrow[8];
    for (int r = 0; r < 8; ++r) { mrow[r] = -3.0e38f; lrow[r] = 0.f; }

    for (int kb = 0; kb < SEQ / 64; ++kb) {
        __syncthreads();   // previous tile consumed
        // ---- K tile via TDM; V transposed manually (TDM cannot transpose);
        //      the vector loads of V overlap the K-tile DMA.
        if (wid == 0) {
            tdm_load_tile((unsigned)(size_t)(void*)&Ks[0],
                          K + (rowbase + kb * 64) * DMODEL + cbase,
                          DMODEL, NTOK, DMODEL, 64, 64, /*padI=*/4, /*padA=*/3);
        }
        uint4 fv[4];
        for (int i = 0; i < 4; ++i) {                 // batch loads first (MLP)
            int idx = (i * 128 + t) * 8;
            int r = idx >> 6, c = idx & 63;
            fv[i] = *(const uint4*)(V + (rowbase + kb * 64 + r) * DMODEL + cbase + c);
        }
        for (int i = 0; i < 4; ++i) {
            int idx = (i * 128 + t) * 8;
            int r = idx >> 6, c = idx & 63;
            const __bf16* hv = (const __bf16*)&fv[i];
            for (int j = 0; j < 8; ++j) Vt[(c + j) * ALD + r] = hv[j];
        }
        if (wid == 0) __builtin_amdgcn_s_wait_tensorcnt(0);
        __syncthreads();

        // ---- S = (Q K^T) * 1/sqrt(depth)
        v8f s[4];
        for (int n = 0; n < 4; ++n) {
            s[n] = vzero;
            v16bf b0 = load_frag_b(Ks, ALD, n * 16, 0);
            v16bf b1 = load_frag_b(Ks, ALD, n * 16, 32);
            s[n] = wmma_bf16(qa0, b0, s[n]);
            s[n] = wmma_bf16(qa1, b1, s[n]);
            for (int r = 0; r < 8; ++r) s[n][r] *= 0.125f;
        }

        // ---- online softmax in C-fragment layout
        for (int r = 0; r < 8; ++r) {
            float mx = fmaxf(fmaxf(s[0][r], s[1][r]), fmaxf(s[2][r], s[3][r]));
            for (int off = 1; off < 16; off <<= 1)
                mx = fmaxf(mx, __shfl_xor(mx, off, 32));
            float mnew  = fmaxf(mrow[r], mx);
            float alpha = __expf(mrow[r] - mnew);
            mrow[r] = mnew;
            float ssum = 0.f;
            for (int n = 0; n < 4; ++n) {
                float p = __expf(s[n][r] - mnew);
                s[n][r] = p;
                ssum += p;
            }
            for (int off = 1; off < 16; off <<= 1)
                ssum += __shfl_xor(ssum, off, 32);
            lrow[r] = lrow[r] * alpha + ssum;
            for (int n = 0; n < 4; ++n) oacc[n][r] *= alpha;
        }

        // ---- C-layout -> A-layout via per-wave LDS patch
        {
            __bf16* pp = &Ps[wid * 16 * ALD];
            const int cn = lane & 15;
            const int hi = lane >> 4;
            for (int n = 0; n < 4; ++n)
                for (int r = 0; r < 8; ++r)
                    pp[(r + 8 * hi) * ALD + n * 16 + cn] = (__bf16)s[n][r];
        }
        asm volatile("s_wait_dscnt 0" ::: "memory");

        v16bf pa0 = load_frag_a(&Ps[wid * 16 * ALD], ALD, 0, 0);
        v16bf pa1 = load_frag_a(&Ps[wid * 16 * ALD], ALD, 0, 32);
        for (int n = 0; n < 4; ++n) {
            v16bf vb0 = load_frag_b(Vt, ALD, n * 16, 0);
            v16bf vb1 = load_frag_b(Vt, ALD, n * 16, 32);
            oacc[n] = wmma_bf16(pa0, vb0, oacc[n]);
            oacc[n] = wmma_bf16(pa1, vb1, oacc[n]);
        }
    }

    // ---- normalize, store bf16
    const int cn = lane & 15;
    const int hi = lane >> 4;
    for (int r = 0; r < 8; ++r) {
        float inv = 1.0f / lrow[r];
        int qrow = qblk * 64 + wid * 16 + r + 8 * hi;
        for (int n = 0; n < 4; ++n)
            O[(rowbase + qrow) * DMODEL + cbase + n * 16 + cn] =
                (__bf16)(oacc[n][r] * inv);
    }
}

// ---------------------------------------------------------------------------
// Residual add + LayerNorm over D=1024; writes fp32 + bf16 copies.
// ---------------------------------------------------------------------------
__global__ __launch_bounds__(256)
void add_ln_kernel(const float* __restrict__ X, const float* __restrict__ R,
                   const float* __restrict__ g, const float* __restrict__ be,
                   float* __restrict__ Yf, __bf16* __restrict__ Yb) {
    __shared__ float sbuf[8];
    const int t = threadIdx.x;
    const size_t row = blockIdx.x;
    float y[4];
    float s = 0.f;
    for (int i = 0; i < 4; ++i) {
        int d = t + i * 256;
        y[i] = X[row * DMODEL + d] + R[row * DMODEL + d];
        s += y[i];
    }
    for (int off = 16; off > 0; off >>= 1) s += __shfl_xor(s, off, 32);
    if ((t & 31) == 0) sbuf[t >> 5] = s;
    __syncthreads();
    if (t == 0) { float a = 0.f; for (int i = 0; i < 8; ++i) a += sbuf[i]; sbuf[0] = a; }
    __syncthreads();
    float mean = sbuf[0] * (1.0f / DMODEL);
    __syncthreads();

    float vs = 0.f;
    for (int i = 0; i < 4; ++i) { float d = y[i] - mean; vs += d * d; }
    for (int off = 16; off > 0; off >>= 1) vs += __shfl_xor(vs, off, 32);
    if ((t & 31) == 0) sbuf[t >> 5] = vs;
    __syncthreads();
    if (t == 0) { float a = 0.f; for (int i = 0; i < 8; ++i) a += sbuf[i]; sbuf[0] = a; }
    __syncthreads();
    float rstd = rsqrtf(sbuf[0] * (1.0f / DMODEL) + EPS);

    for (int i = 0; i < 4; ++i) {
        int d = t + i * 256;
        float o = (y[i] - mean) * rstd * g[d] + be[d];
        Yf[row * DMODEL + d] = o;
        Yb[row * DMODEL + d] = (__bf16)o;
    }
}

// ---------------------------------------------------------------------------
// Embedding * sqrt(D) + positional encoding; fp32 + bf16 copies.
// ---------------------------------------------------------------------------
__global__ __launch_bounds__(256)
void embed_kernel(const int* __restrict__ tokens, const float* __restrict__ emb,
                  float* __restrict__ Xf, __bf16* __restrict__ Xb) {
    const size_t row = blockIdx.x;
    const int s = (int)(row % SEQ);
    const int tok = tokens[row];
    const int t = threadIdx.x;
    const float kln = -logf(10000.0f) / (float)DMODEL;
    for (int i = 0; i < 4; ++i) {
        int d = t + i * 256;
        float e = emb[(size_t)tok * DMODEL + d] * 32.0f;   // sqrt(1024)
        int pair = d >> 1;
        float div = __expf((float)(2 * pair) * kln);
        float ang = (float)s * div;
        float pe = (d & 1) ? __cosf(ang) : __sinf(ang);
        float o = e + pe;
        Xf[row * DMODEL + d] = o;
        Xb[row * DMODEL + d] = (__bf16)o;
    }
}

// ---------------------------------------------------------------------------
// In-place row softmax over VOCAB.
// ---------------------------------------------------------------------------
__global__ __launch_bounds__(256)
void softmax_rows_kernel(float* __restrict__ X, int n) {
    __shared__ float sbuf[8];
    const int t = threadIdx.x;
    float* row = X + (size_t)blockIdx.x * n;

    float mx = -3.0e38f;
    for (int i = t; i < n; i += 256) mx = fmaxf(mx, row[i]);
    for (int off = 16; off > 0; off >>= 1) mx = fmaxf(mx, __shfl_xor(mx, off, 32));
    if ((t & 31) == 0) sbuf[t >> 5] = mx;
    __syncthreads();
    if (t == 0) { float a = sbuf[0]; for (int i = 1; i < 8; ++i) a = fmaxf(a, sbuf[i]); sbuf[0] = a; }
    __syncthreads();
    mx = sbuf[0];
    __syncthreads();

    float s = 0.f;
    for (int i = t; i < n; i += 256) {
        float p = __expf(row[i] - mx);
        row[i] = p;
        s += p;
    }
    for (int off = 16; off > 0; off >>= 1) s += __shfl_xor(s, off, 32);
    if ((t & 31) == 0) sbuf[t >> 5] = s;
    __syncthreads();
    if (t == 0) { float a = 0.f; for (int i = 0; i < 8; ++i) a += sbuf[i]; sbuf[0] = a; }
    __syncthreads();
    float inv = 1.0f / sbuf[0];
    for (int i = t; i < n; i += 256) row[i] *= inv;
}

// ---------------------------------------------------------------------------
// Host orchestration
// ---------------------------------------------------------------------------
static inline void launch_gemm(const __bf16* A, const __bf16* Wt, const float* bias,
                               float* Cf, __bf16* Cb, int M, int N, int K,
                               int relu, hipStream_t stream) {
    dim3 grid(N / GBN, M / GBM);
    gemm_tdm_kernel<<<grid, 256, 0, stream>>>(A, Wt, bias, Cf, Cb, M, N, K, relu);
}

static inline void launch_transpose(const float* W, __bf16* Wt, int K, int N,
                                    hipStream_t stream) {
    transpose_bf16_kernel<<<dim3(N / 32, K / 32), 256, 0, stream>>>(W, Wt, K, N);
}

extern "C" void kernel_launch(void* const* d_in, const int* in_sizes, int n_in,
                              void* d_out, int out_size, void* d_ws, size_t ws_size,
                              hipStream_t stream) {
    const int*   tokens = (const int*)  d_in[0];
    const float* emb    = (const float*)d_in[1];
    const float* Wq     = (const float*)d_in[2];
    const float* bq     = (const float*)d_in[3];
    const float* Wk     = (const float*)d_in[4];
    const float* bk     = (const float*)d_in[5];
    const float* Wv     = (const float*)d_in[6];
    const float* bv     = (const float*)d_in[7];
    const float* Wo     = (const float*)d_in[8];
    const float* bo     = (const float*)d_in[9];
    const float* W1     = (const float*)d_in[10];
    const float* b1     = (const float*)d_in[11];
    const float* W2     = (const float*)d_in[12];
    const float* b2     = (const float*)d_in[13];
    const float* ln1g   = (const float*)d_in[14];
    const float* ln1b   = (const float*)d_in[15];
    const float* ln2g   = (const float*)d_in[16];
    const float* ln2b   = (const float*)d_in[17];
    const float* Wf     = (const float*)d_in[18];
    const float* bfv    = (const float*)d_in[19];

    // ---- workspace layout (256B-aligned chunks)
    char* wp = (char*)d_ws;
    auto alloc = [&](size_t bytes) {
        void* r = (void*)wp;
        wp += (bytes + 255) & ~(size_t)255;
        return r;
    };
    const size_t TD = (size_t)NTOK * DMODEL;
    float*  xf    = (float*) alloc(TD * 4);
    __bf16* xb    = (__bf16*)alloc(TD * 2);
    float*  x1f   = (float*) alloc(TD * 4);
    __bf16* x1b   = (__bf16*)alloc(TD * 2);
    __bf16* qb    = (__bf16*)alloc(TD * 2);
    __bf16* kb    = (__bf16*)alloc(TD * 2);
    __bf16* vb    = (__bf16*)alloc(TD * 2);
    __bf16* ob    = (__bf16*)alloc(TD * 2);
    float*  projf = (float*) alloc(TD * 4);
    __bf16* hb    = (__bf16*)alloc((size_t)NTOK * DFFN * 2);
    const size_t WDD = (size_t)DMODEL * DMODEL;
    const size_t WDF = (size_t)DMODEL * DFFN;
    __bf16* wqt = (__bf16*)alloc(LAYERS * WDD * 2);
    __bf16* wkt = (__bf16*)alloc(LAYERS * WDD * 2);
    __bf16* wvt = (__bf16*)alloc(LAYERS * WDD * 2);
    __bf16* wot = (__bf16*)alloc(LAYERS * WDD * 2);
    __bf16* w1t = (__bf16*)alloc(LAYERS * WDF * 2);
    __bf16* w2t = (__bf16*)alloc(LAYERS * WDF * 2);
    __bf16* wft = (__bf16*)alloc((size_t)DMODEL * VOCAB * 2);
    float* logits = (float*)d_out;

    // ---- weight prep: fp32 -> bf16, pre-transposed to [N][K]
    for (int l = 0; l < LAYERS; ++l) {
        launch_transpose(Wq + l * WDD, wqt + l * WDD, DMODEL, DMODEL, stream);
        launch_transpose(Wk + l * WDD, wkt + l * WDD, DMODEL, DMODEL, stream);
        launch_transpose(Wv + l * WDD, wvt + l * WDD, DMODEL, DMODEL, stream);
        launch_transpose(Wo + l * WDD, wot + l * WDD, DMODEL, DMODEL, stream);
        launch_transpose(W1 + l * WDF, w1t + l * WDF, DMODEL, DFFN, stream);
        launch_transpose(W2 + l * WDF, w2t + l * WDF, DFFN, DMODEL, stream);
    }
    launch_transpose(Wf, wft, DMODEL, VOCAB, stream);

    // ---- embedding + positional encoding
    embed_kernel<<<NTOK, 256, 0, stream>>>(tokens, emb, xf, xb);

    for (int l = 0; l < LAYERS; ++l) {
        const size_t oDD = l * WDD, oDF = l * WDF;
        const size_t bD = (size_t)l * DMODEL, bF = (size_t)l * DFFN;

        launch_gemm(xb, wqt + oDD, bq + bD, nullptr, qb, NTOK, DMODEL, DMODEL, 0, stream);
        launch_gemm(xb, wkt + oDD, bk + bD, nullptr, kb, NTOK, DMODEL, DMODEL, 0, stream);
        launch_gemm(xb, wvt + oDD, bv + bD, nullptr, vb, NTOK, DMODEL, DMODEL, 0, stream);

        attention_kernel<<<dim3(SEQ / 64, BATCH * NHEAD), 128, 0, stream>>>(qb, kb, vb, ob);

        launch_gemm(ob, wot + oDD, bo + bD, projf, nullptr, NTOK, DMODEL, DMODEL, 0, stream);
        add_ln_kernel<<<NTOK, 256, 0, stream>>>(xf, projf, ln1g + bD, ln1b + bD, x1f, x1b);

        launch_gemm(x1b, w1t + oDF, b1 + bF, nullptr, hb, NTOK, DFFN, DMODEL, 1, stream);
        launch_gemm(hb, w2t + oDF, b2 + bD, projf, nullptr, NTOK, DMODEL, DFFN, 0, stream);
        add_ln_kernel<<<NTOK, 256, 0, stream>>>(x1f, projf, ln2g + bD, ln2b + bD, xf, xb);
    }

    // ---- final projection + softmax (in place in d_out)
    launch_gemm(xb, wft, bfv, logits, nullptr, NTOK, VOCAB, DMODEL, 0, stream);
    softmax_rows_kernel<<<NTOK, 256, 0, stream>>>(logits, VOCAB);
}
